// CheckersGPT_3058016714916
// MI455X (gfx1250) — compile-verified, matrix operand order checked by reading, not run on hardware
//
#include <hip/hip_runtime.h>

// ---------------------------------------------------------------------------
// CheckersGPT forward on MI455X (gfx1250).
// Compute-bound (~0.54 TFLOP/fwd) -> all big GEMMs on v_wmma_f32_16x16x32_bf16.
// LDS-staged 128x128 block tiles, double-buffered K=32 slabs, async
// global->LDS copies (GLOBAL_LOAD_ASYNC_TO_LDS_B128, ASYNCcnt).
// ---------------------------------------------------------------------------

typedef __bf16 bf16_t;
typedef __attribute__((ext_vector_type(16))) __bf16 v16bf;
typedef __attribute__((ext_vector_type(8)))  __bf16 v8bf;
typedef __attribute__((ext_vector_type(8)))  float  v8f;

#define VSZ 512
#define ESZ 512
#define LSZ 6
#define HSZ 8
#define BSZ 16
#define TSZ 256
#define MROWS (BSZ * TSZ)   // 4096
#define HE    (HSZ * ESZ)   // 4096

#define BLK_M 128
#define BLK_N 128
#define BLK_K 32
#define LDS_PITCH 40        // 32 elems + 8 pad: 16 consecutive rows hit 16 banks

#if defined(__AMDGCN__) && __has_builtin(__builtin_amdgcn_global_load_async_to_lds_b128)
#define HAVE_ASYNC_LDS 1
#else
#define HAVE_ASYNC_LDS 0
#endif

// Builtin parameter types (from hipcc diagnostic): pointer to int __vector(4)
// in global (1) / LDS (3) address spaces.
typedef int v4i_t __attribute__((vector_size(16)));
typedef __attribute__((address_space(1))) v4i_t* gptr_v4i;
typedef __attribute__((address_space(3))) v4i_t* lptr_v4i;

__device__ __forceinline__ void async_copy16(const bf16_t* g, bf16_t* s) {
#if HAVE_ASYNC_LDS
  // 16B global -> LDS, tracked by ASYNCcnt (GLOBAL_LOAD_ASYNC_TO_LDS_B128).
  // Integer detour: strips const, and truncating the generic LDS address to
  // 32 bits yields the LDS offset (ISA 10.2).
  __builtin_amdgcn_global_load_async_to_lds_b128(
      (gptr_v4i)(unsigned long long)g,
      (lptr_v4i)(unsigned int)(unsigned long long)s, 0, 0);
#else
  *reinterpret_cast<v8bf*>(s) = *reinterpret_cast<const v8bf*>(g);
#endif
}

__device__ __forceinline__ void wait_async0() {
#if HAVE_ASYNC_LDS
#if __has_builtin(__builtin_amdgcn_s_wait_asynccnt)
  __builtin_amdgcn_s_wait_asynccnt(0);
#else
  asm volatile("s_wait_asynccnt 0x0" ::: "memory");
#endif
#endif
}

// ---------------------------------------------------------------------------
// Embedding + sinusoidal PE (concatenated sin|cos as in reference)
// ---------------------------------------------------------------------------
__global__ void embed_pe_kernel(const int* __restrict__ toks,
                                const float* __restrict__ emb,
                                float* __restrict__ xf, bf16_t* __restrict__ xh) {
  size_t idx = (size_t)blockIdx.x * blockDim.x + threadIdx.x;
  if (idx >= (size_t)MROWS * ESZ) return;
  int e  = (int)(idx % ESZ);
  int bt = (int)(idx / ESZ);
  int t  = bt % TSZ;
  int tok = toks[bt];
  float pe;
  if (e < ESZ / 2) {
    float rate = __powf(10000.0f, -2.0f * (float)e / (float)ESZ);
    pe = __sinf((float)t * rate);
  } else {
    int c = e - ESZ / 2;
    float rate = __powf(10000.0f, -2.0f * (float)c / (float)ESZ);
    pe = __cosf((float)t * rate);
  }
  float v = emb[(size_t)tok * ESZ + e] + pe;
  xf[idx] = v;
  xh[idx] = (bf16_t)v;
}

// ---------------------------------------------------------------------------
// f32 [batch,R,C] -> bf16 [batch,C,R]  (weight pack: B stored as [N,K])
// ---------------------------------------------------------------------------
__global__ void conv_transpose_kernel(const float* __restrict__ in,
                                      bf16_t* __restrict__ out,
                                      int R, int C, int batch) {
  size_t total = (size_t)batch * R * C;
  size_t idx = (size_t)blockIdx.x * blockDim.x + threadIdx.x;
  if (idx >= total) return;
  size_t rc  = (size_t)R * C;
  int b = (int)(idx / rc);
  size_t rem = idx - (size_t)b * rc;
  int r = (int)(rem / C);
  int c = (int)(rem % C);
  out[(size_t)b * rc + (size_t)c * R + r] = (bf16_t)in[idx];
}

// bf16 V [B*T, E] -> bf16 Vt [B, E, T]
__global__ void transpose_v_kernel(const bf16_t* __restrict__ V,
                                   bf16_t* __restrict__ Vt) {
  size_t idx = (size_t)blockIdx.x * blockDim.x + threadIdx.x;
  if (idx >= (size_t)MROWS * ESZ) return;
  int e  = (int)(idx % ESZ);
  int bt = (int)(idx / ESZ);
  int b = bt / TSZ, t = bt % TSZ;
  Vt[((size_t)b * ESZ + e) * TSZ + t] = V[idx];
}

// ---------------------------------------------------------------------------
// bf16 WMMA GEMM: C = A[M,K] * Bp[N,K]^T (+bias)(+relu)
// Block: 256 thr (8 waves), 128x128 tile; wave tile 32x64 (2x4 accumulators).
// K staged through LDS in 32-wide slabs, double-buffered, async copies.
// Fragment layout per ISA 7.12.2 (16-bit A/B 16x32): lane<16 holds K 0..7 &
// 16..23; lane>=16 holds K 8..15 & 24..31 -> two 16B reads per fragment.
// ---------------------------------------------------------------------------
__global__ void __launch_bounds__(256)
gemm_bf16_wmma(const bf16_t* __restrict__ A, long long lda, long long sA,
               const bf16_t* __restrict__ Bp, long long ldb, long long sB,
               float* __restrict__ Cf, bf16_t* __restrict__ Ch,
               long long ldc, long long sC,
               const float* __restrict__ bias, int relu,
               int M, int N, int K) {
  __shared__ bf16_t Asm[2][BLK_M * LDS_PITCH];
  __shared__ bf16_t Bsm[2][BLK_N * LDS_PITCH];

  int tid  = threadIdx.x;
  int lane = tid & 31;
  int wave = tid >> 5;
  int tilesN = N / BLK_N;
  int bm = (blockIdx.x / tilesN) * BLK_M;
  int bn = (blockIdx.x % tilesN) * BLK_N;
  int half = lane >> 4;
  int l15  = lane & 15;
  int m0w = (wave >> 1) * 32;   // 0,32,64,96
  int n0w = (wave & 1) * 64;    // 0,64

  const bf16_t* Ab = A  + (size_t)blockIdx.y * sA;
  const bf16_t* Bb = Bp + (size_t)blockIdx.y * sB;

  // staging: 512 x 16B chunks per matrix, 2 per thread (chunk c: row c>>2,
  // elem offset (c&3)*8)
  int ca0 = tid, ca1 = tid + 256;

  v8f acc[2][4];
#pragma unroll
  for (int i = 0; i < 2; ++i)
#pragma unroll
    for (int j = 0; j < 4; ++j)
#pragma unroll
      for (int e = 0; e < 8; ++e) acc[i][j][e] = 0.0f;

  int nk = K / BLK_K;

  // prologue: stage slab 0 into buffer 0
  {
    int r0 = ca0 >> 2, o0 = (ca0 & 3) * 8;
    int r1 = ca1 >> 2, o1 = (ca1 & 3) * 8;
    async_copy16(Ab + (size_t)(bm + r0) * lda + o0, &Asm[0][r0 * LDS_PITCH + o0]);
    async_copy16(Ab + (size_t)(bm + r1) * lda + o1, &Asm[0][r1 * LDS_PITCH + o1]);
    async_copy16(Bb + (size_t)(bn + r0) * ldb + o0, &Bsm[0][r0 * LDS_PITCH + o0]);
    async_copy16(Bb + (size_t)(bn + r1) * ldb + o1, &Bsm[0][r1 * LDS_PITCH + o1]);
  }

  for (int it = 0; it < nk; ++it) {
    int cur = it & 1;
    wait_async0();
    __syncthreads();
    if (it + 1 < nk) {
      int k0 = (it + 1) * BLK_K;
      int nxt = cur ^ 1;
      int r0 = ca0 >> 2, o0 = (ca0 & 3) * 8;
      int r1 = ca1 >> 2, o1 = (ca1 & 3) * 8;
      async_copy16(Ab + (size_t)(bm + r0) * lda + k0 + o0, &Asm[nxt][r0 * LDS_PITCH + o0]);
      async_copy16(Ab + (size_t)(bm + r1) * lda + k0 + o1, &Asm[nxt][r1 * LDS_PITCH + o1]);
      async_copy16(Bb + (size_t)(bn + r0) * ldb + k0 + o0, &Bsm[nxt][r0 * LDS_PITCH + o0]);
      async_copy16(Bb + (size_t)(bn + r1) * ldb + k0 + o1, &Bsm[nxt][r1 * LDS_PITCH + o1]);
    }
    // fragments from LDS, then 8 WMMAs
    const bf16_t* As = &Asm[cur][0];
    const bf16_t* Bs = &Bsm[cur][0];
    v16bf af[2], bfr[4];
#pragma unroll
    for (int i = 0; i < 2; ++i) {
      const bf16_t* p = As + (m0w + 16 * i + l15) * LDS_PITCH + half * 8;
      v8bf lo = *reinterpret_cast<const v8bf*>(p);
      v8bf hi = *reinterpret_cast<const v8bf*>(p + 16);
      af[i] = __builtin_shufflevector(lo, hi, 0, 1, 2, 3, 4, 5, 6, 7,
                                      8, 9, 10, 11, 12, 13, 14, 15);
    }
#pragma unroll
    for (int j = 0; j < 4; ++j) {
      const bf16_t* q = Bs + (n0w + 16 * j + l15) * LDS_PITCH + half * 8;
      v8bf lo = *reinterpret_cast<const v8bf*>(q);
      v8bf hi = *reinterpret_cast<const v8bf*>(q + 16);
      bfr[j] = __builtin_shufflevector(lo, hi, 0, 1, 2, 3, 4, 5, 6, 7,
                                       8, 9, 10, 11, 12, 13, 14, 15);
    }
#pragma unroll
    for (int i = 0; i < 2; ++i)
#pragma unroll
      for (int j = 0; j < 4; ++j)
        acc[i][j] = __builtin_amdgcn_wmma_f32_16x16x32_bf16(
            false, af[i], false, bfr[j], (short)0, acc[i][j], false, false);
  }

  float*  Cfb = Cf ? Cf + (size_t)blockIdx.y * sC : nullptr;
  bf16_t* Chb = Ch ? Ch + (size_t)blockIdx.y * sC : nullptr;
#pragma unroll
  for (int i = 0; i < 2; ++i) {
#pragma unroll
    for (int j = 0; j < 4; ++j) {
      int n = bn + n0w + 16 * j + l15;
      float bv = bias ? bias[n] : 0.0f;
#pragma unroll
      for (int vg = 0; vg < 8; ++vg) {
        int m = bm + m0w + 16 * i + vg + 8 * half;   // C layout (ISA 7.12.2)
        float v = acc[i][j][vg] + bv;
        if (relu) v = fmaxf(v, 0.0f);
        if (Cfb) Cfb[(size_t)m * ldc + n] = v;
        if (Chb) Chb[(size_t)m * ldc + n] = (bf16_t)v;
      }
    }
  }
}

// ---------------------------------------------------------------------------
// Causal softmax over energy rows [B*T, T] -> bf16 P
// ---------------------------------------------------------------------------
__global__ void softmax_causal_kernel(const float* __restrict__ En,
                                      bf16_t* __restrict__ P) {
  __shared__ float red[256];
  int r = blockIdx.x;       // row over B*T
  int i = r % TSZ;          // query position
  int j = threadIdx.x;      // key position (T == 256 == blockDim)
  float v = (j <= i) ? En[(size_t)r * TSZ + j] : -1e9f;
  red[j] = v;
  __syncthreads();
  for (int s = 128; s > 0; s >>= 1) {
    if (j < s) red[j] = fmaxf(red[j], red[j + s]);
    __syncthreads();
  }
  float mx = red[0];
  __syncthreads();
  float ex = __expf(v - mx);
  red[j] = ex;
  __syncthreads();
  for (int s = 128; s > 0; s >>= 1) {
    if (j < s) red[j] += red[j + s];
    __syncthreads();
  }
  float inv = 1.0f / red[0];
  P[(size_t)r * TSZ + j] = (bf16_t)(ex * inv);
}

// ---------------------------------------------------------------------------
// y = LayerNorm(a + b) * w + bb ; row-wise over E=512, 256 thr x 2 elems
// ---------------------------------------------------------------------------
__global__ void add_ln_kernel(const float* __restrict__ a,
                              const float* __restrict__ b,
                              const float* __restrict__ w,
                              const float* __restrict__ bb,
                              float* __restrict__ yf, bf16_t* __restrict__ yh) {
  __shared__ float s1[256], s2[256];
  int r = blockIdx.x, tid = threadIdx.x;
  const float* ar = a + (size_t)r * ESZ;
  const float* br = b + (size_t)r * ESZ;
  float x0 = ar[tid] + br[tid];
  float x1 = ar[tid + 256] + br[tid + 256];
  s1[tid] = x0 + x1;
  s2[tid] = x0 * x0 + x1 * x1;
  __syncthreads();
  for (int s = 128; s > 0; s >>= 1) {
    if (tid < s) { s1[tid] += s1[tid + s]; s2[tid] += s2[tid + s]; }
    __syncthreads();
  }
  float mean = s1[0] * (1.0f / ESZ);
  float var  = s2[0] * (1.0f / ESZ) - mean * mean;
  float inv  = rsqrtf(var + 1e-5f);
  float y0 = (x0 - mean) * inv * w[tid] + bb[tid];
  float y1 = (x1 - mean) * inv * w[tid + 256] + bb[tid + 256];
  size_t o = (size_t)r * ESZ + tid;
  if (yf) { yf[o] = y0; yf[o + 256] = y1; }
  if (yh) { yh[o] = (bf16_t)y0; yh[o + 256] = (bf16_t)y1; }
}

// ---------------------------------------------------------------------------
// Final: logits = x[:, -1, :] @ wout + bout ; softmax -> out [B, V]
// ---------------------------------------------------------------------------
__global__ void logits_softmax_kernel(const float* __restrict__ xf,
                                      const float* __restrict__ wout,
                                      const float* __restrict__ bout,
                                      float* __restrict__ out) {
  __shared__ float xs[ESZ];
  __shared__ float red[256];
  int b = blockIdx.x, tid = threadIdx.x;
  const float* xr = xf + ((size_t)b * TSZ + (TSZ - 1)) * ESZ;
  xs[tid] = xr[tid];
  xs[tid + 256] = xr[tid + 256];
  __syncthreads();
  float l0 = bout[tid], l1 = bout[tid + 256];
  for (int e = 0; e < ESZ; ++e) {
    float xv = xs[e];
    l0 += xv * wout[(size_t)e * VSZ + tid];
    l1 += xv * wout[(size_t)e * VSZ + tid + 256];
  }
  red[tid] = fmaxf(l0, l1);
  __syncthreads();
  for (int s = 128; s > 0; s >>= 1) {
    if (tid < s) red[tid] = fmaxf(red[tid], red[tid + s]);
    __syncthreads();
  }
  float mx = red[0];
  __syncthreads();
  float e0 = __expf(l0 - mx), e1 = __expf(l1 - mx);
  red[tid] = e0 + e1;
  __syncthreads();
  for (int s = 128; s > 0; s >>= 1) {
    if (tid < s) red[tid] += red[tid + s];
    __syncthreads();
  }
  float inv = 1.0f / red[0];
  out[(size_t)b * VSZ + tid] = e0 * inv;
  out[(size_t)b * VSZ + tid + 256] = e1 * inv;
}

// ---------------------------------------------------------------------------
// Host sequencing
// ---------------------------------------------------------------------------
static inline void launch_gemm(hipStream_t st,
                               const bf16_t* A, long long lda, long long sA,
                               const bf16_t* B, long long ldb, long long sB,
                               float* Cf, bf16_t* Ch, long long ldc, long long sC,
                               const float* bias, int relu,
                               int M, int N, int K, int batch) {
  dim3 grid((M / BLK_M) * (N / BLK_N), batch, 1);
  gemm_bf16_wmma<<<grid, 256, 0, st>>>(A, lda, sA, B, ldb, sB,
                                       Cf, Ch, ldc, sC, bias, relu, M, N, K);
}

extern "C" void kernel_launch(void* const* d_in, const int* in_sizes, int n_in,
                              void* d_out, int out_size, void* d_ws, size_t ws_size,
                              hipStream_t stream) {
  (void)in_sizes; (void)n_in; (void)out_size; (void)ws_size;
  const int*   toks  = (const int*)d_in[0];
  const float* emb   = (const float*)d_in[1];
  const float* wq    = (const float*)d_in[2];
  const float* wk    = (const float*)d_in[3];
  const float* wv    = (const float*)d_in[4];
  const float* wo    = (const float*)d_in[5];
  const float* bo    = (const float*)d_in[6];
  const float* ln1w  = (const float*)d_in[7];
  const float* ln1b  = (const float*)d_in[8];
  const float* ln2w  = (const float*)d_in[9];
  const float* ln2b  = (const float*)d_in[10];
  const float* ff1w  = (const float*)d_in[11];
  const float* ff1b  = (const float*)d_in[12];
  const float* ff2w  = (const float*)d_in[13];
  const float* ff2b  = (const float*)d_in[14];
  const float* woutp = (const float*)d_in[15];
  const float* boutp = (const float*)d_in[16];
  float* out = (float*)d_out;
  char*  ws  = (char*)d_ws;

  // workspace layout (bytes)
  float*  xf   = (float*) (ws + 0);                  //  8 MB  [4096,512] f32
  bf16_t* xh   = (bf16_t*)(ws + 8388608);            //  4 MB
  bf16_t* wqT  = (bf16_t*)(ws + 12582912);           //  4 MB  [H,512,512]
  bf16_t* wkT  = (bf16_t*)(ws + 16777216);           //  4 MB
  bf16_t* wvT  = (bf16_t*)(ws + 20971520);           //  4 MB
  bf16_t* woT  = (bf16_t*)(ws + 25165824);           //  4 MB  [512,4096]
  bf16_t* f1T  = (bf16_t*)(ws + 29360128);           // .5 MB  [512,512]
  bf16_t* f2T  = (bf16_t*)(ws + 29884416);           // .5 MB
  bf16_t* Qh   = (bf16_t*)(ws + 30408704);           //  4 MB  [4096,512]
  bf16_t* Kh   = (bf16_t*)(ws + 34603008);           //  4 MB
  bf16_t* Vh   = (bf16_t*)(ws + 38797312);           //  4 MB
  bf16_t* Vt   = (bf16_t*)(ws + 42991616);           //  4 MB  [B,512,256]
  float*  En   = (float*) (ws + 47185920);           //  4 MB  [B,256,256]
  bf16_t* Ph   = (bf16_t*)(ws + 51380224);           //  2 MB
  bf16_t* Ob   = (bf16_t*)(ws + 53477376);           // 32 MB  [4096,4096]
  float*  attn = (float*) (ws + 87031808);           //  8 MB  [4096,512]
  bf16_t* h1h  = (bf16_t*)(ws + 95420416);           //  4 MB
  bf16_t* f1h  = (bf16_t*)(ws + 99614720);           //  4 MB
  float*  f2f  = (float*) (ws + 103809024);          //  8 MB

  const int TPB = 256;
  size_t nXE = (size_t)MROWS * ESZ;

  embed_pe_kernel<<<(nXE + TPB - 1) / TPB, TPB, 0, stream>>>(toks, emb, xf, xh);

  for (int l = 0; l < LSZ; ++l) {
    size_t lqkv = (size_t)l * HSZ * ESZ * ESZ;
    size_t lff  = (size_t)l * ESZ * ESZ;
    // pack this layer's weights to bf16 [N,K]
    size_t nW = (size_t)HSZ * ESZ * ESZ;
    conv_transpose_kernel<<<(nW + TPB - 1) / TPB, TPB, 0, stream>>>(wq + lqkv, wqT, ESZ, ESZ, HSZ);
    conv_transpose_kernel<<<(nW + TPB - 1) / TPB, TPB, 0, stream>>>(wk + lqkv, wkT, ESZ, ESZ, HSZ);
    conv_transpose_kernel<<<(nW + TPB - 1) / TPB, TPB, 0, stream>>>(wv + lqkv, wvT, ESZ, ESZ, HSZ);
    size_t nWo = (size_t)HE * ESZ;
    conv_transpose_kernel<<<(nWo + TPB - 1) / TPB, TPB, 0, stream>>>(wo + (size_t)l * HE * ESZ, woT, HE, ESZ, 1);
    size_t nF = (size_t)ESZ * ESZ;
    conv_transpose_kernel<<<(nF + TPB - 1) / TPB, TPB, 0, stream>>>(ff1w + lff, f1T, ESZ, ESZ, 1);
    conv_transpose_kernel<<<(nF + TPB - 1) / TPB, TPB, 0, stream>>>(ff2w + lff, f2T, ESZ, ESZ, 1);

    for (int h = 0; h < HSZ; ++h) {
      const bf16_t* wqh = wqT + (size_t)h * ESZ * ESZ;
      const bf16_t* wkh = wkT + (size_t)h * ESZ * ESZ;
      const bf16_t* wvh = wvT + (size_t)h * ESZ * ESZ;
      // Q/K/V = x @ W  -> bf16 [4096,512]
      launch_gemm(stream, xh, ESZ, 0, wqh, ESZ, 0, nullptr, Qh, ESZ, 0,
                  nullptr, 0, MROWS, ESZ, ESZ, 1);
      launch_gemm(stream, xh, ESZ, 0, wkh, ESZ, 0, nullptr, Kh, ESZ, 0,
                  nullptr, 0, MROWS, ESZ, ESZ, 1);
      launch_gemm(stream, xh, ESZ, 0, wvh, ESZ, 0, nullptr, Vh, ESZ, 0,
                  nullptr, 0, MROWS, ESZ, ESZ, 1);
      transpose_v_kernel<<<(nXE + TPB - 1) / TPB, TPB, 0, stream>>>(Vh, Vt);
      // energy[b] = Q_b @ K_b^T : B-packed == K rows directly (K-dim = E)
      launch_gemm(stream, Qh, ESZ, (long long)TSZ * ESZ,
                  Kh, ESZ, (long long)TSZ * ESZ,
                  En, nullptr, TSZ, (long long)TSZ * TSZ,
                  nullptr, 0, TSZ, TSZ, ESZ, BSZ);
      softmax_causal_kernel<<<MROWS, TSZ, 0, stream>>>(En, Ph);
      // O[b] = P_b @ V_b : B-packed = Vt[b] [E,T]; write into Ob at col h*E
      launch_gemm(stream, Ph, TSZ, (long long)TSZ * TSZ,
                  Vt, TSZ, (long long)ESZ * TSZ,
                  nullptr, Ob + (size_t)h * ESZ, HE, (long long)TSZ * HE,
                  nullptr, 0, TSZ, ESZ, TSZ, BSZ);
    }
    // attn_out = O @ wo + bo  (f32)
    launch_gemm(stream, Ob, HE, 0, woT, HE, 0, attn, nullptr, ESZ, 0,
                bo + (size_t)l * ESZ, 0, MROWS, ESZ, HE, 1);
    // h1 = LN(attn + x)
    add_ln_kernel<<<MROWS, TPB, 0, stream>>>(attn, xf,
        ln1w + (size_t)l * ESZ, ln1b + (size_t)l * ESZ, nullptr, h1h);
    // ff = relu(h1 @ ff1 + b1) @ ff2 + b2
    launch_gemm(stream, h1h, ESZ, 0, f1T, ESZ, 0, nullptr, f1h, ESZ, 0,
                ff1b + (size_t)l * ESZ, 1, MROWS, ESZ, ESZ, 1);
    launch_gemm(stream, f1h, ESZ, 0, f2T, ESZ, 0, f2f, nullptr, ESZ, 0,
                ff2b + (size_t)l * ESZ, 0, MROWS, ESZ, ESZ, 1);
    // x = LN(attn + ff)
    add_ln_kernel<<<MROWS, TPB, 0, stream>>>(attn, f2f,
        ln2w + (size_t)l * ESZ, ln2b + (size_t)l * ESZ, xf, xh);
  }

  logits_softmax_kernel<<<BSZ, TPB, 0, stream>>>(xf, woutp, boutp, out);
}
